// Decoder_5334349382400
// MI455X (gfx1250) — matrix operand order repaired and checked by reading, not compiled, not run on hardware
//
#include <hip/hip_runtime.h>
#include <hip/hip_bf16.h>

// ---------------------------------------------------------------------------
// Transformer decoder (3 layers, HID=512, 8 heads, Lq=Lk=2048) on gfx1250.
// All GEMMs run through V_WMMA_F32_16X16X32_BF16 (f32 accumulate).
// Wave tile: 32 rows x 64 cols = 8 WMMA accumulators; 24 B/lane of L2
// traffic per WMMA; register double-buffered k-pipeline.
// ---------------------------------------------------------------------------

#define LSEQ   2048
#define HIDDIM 512
#define PFDIM  2048
#define NHEADS 8
#define HDIM   64
#define NLAYER 3

typedef __attribute__((ext_vector_type(16))) __bf16         v16bf;
typedef __attribute__((ext_vector_type(16))) unsigned short v16u;
typedef __attribute__((ext_vector_type(8)))  float          v8f;

__device__ __forceinline__ unsigned short f32_to_bf16_bits(float f) {
    union { float f; unsigned u; } a; a.f = f;
    unsigned r = (a.u + 0x7FFFu + ((a.u >> 16) & 1u)) >> 16;   // round-nearest-even
    return (unsigned short)r;
}

// ---------------------------------------------------------------------------
// Fragment loaders (contiguous-k layouts, wave32). All sources are bf16.
// A: 16x32.  lanes 0-15 : row m0+lane,    K = k0+0..15 contiguous
//            lanes 16-31: row m0+lane-16, K = k0+16..31 contiguous
//    -> one 32B contiguous run: two b128 loads.
// B: 32x16 from Bt[N][K] row-major (B transposed).
//    lanes 0-15 : col n0+lane,    K = k0+0..15 contiguous
//    lanes 16-31: col n0+lane-16, K = k0+16..31 contiguous
//    -> one 32B contiguous run: two b128 loads.
// ---------------------------------------------------------------------------
union FragCast { uint4 q[2]; v16u u; v16bf b; };

__device__ __forceinline__ v16bf load_a_frag(const unsigned short* __restrict__ A, int lda,
                                             int m0, int k0, int lane) {
    int row = m0 + (lane & 15);
    int kb  = k0 + ((lane & 16) ? 16 : 0);
    const unsigned short* p = A + row * lda + kb;
    FragCast f;
    f.q[0] = *(const uint4*)(p);        // K kb .. kb+7
    f.q[1] = *(const uint4*)(p + 8);    // K kb+8 .. kb+15
    return f.b;
}

__device__ __forceinline__ v16bf load_b_frag(const unsigned short* __restrict__ Bt, int ldb,
                                             int n0, int k0, int lane) {
    int col = n0 + (lane & 15);
    int kb  = k0 + ((lane & 16) ? 16 : 0);
    FragCast f;
    f.u = *(const v16u*)(Bt + col * ldb + kb);
    return f.b;
}

// ---------------------------------------------------------------------------
// GEMM: C[M,N] = A[M,K] (bf16) @ Bt[N,K]^T (bf16) (+bias) (+relu)
// Block = 256 threads = 8 waves. Wave w owns rows [m0+32w, +32) x 64 cols:
// 2 A fragments x 4 B fragments -> 8 accumulators. Grid = (N/64, M/256).
// REQUIRES: N % 64 == 0, K % 32 == 0, M % 256 == 0 (true at every call site).
// flags: 1 = relu, 2 = bf16 output, 4 = transposed store C[n*ldc + m]
// ---------------------------------------------------------------------------
#define GF_RELU  1
#define GF_BF16  2
#define GF_TRANS 4

#define WMMA_BF16(a, b, c) \
    __builtin_amdgcn_wmma_f32_16x16x32_bf16(false, (a), false, (b), (short)0, (c), false, false)

__global__ __launch_bounds__(256)
void gemm_wmma(const unsigned short* __restrict__ A, int lda,
               const unsigned short* __restrict__ Bt, int ldb,
               const float* __restrict__ bias,
               void* __restrict__ C, int ldc,
               int M, int N, int K, int flags) {
    int lane = threadIdx.x & 31;
    int wave = threadIdx.x >> 5;
    int m0 = blockIdx.y * 256 + wave * 32;
    int n0 = blockIdx.x * 64;
    if (m0 >= M) return;

    v8f acc00 = {}, acc01 = {}, acc02 = {}, acc03 = {};
    v8f acc10 = {}, acc11 = {}, acc12 = {}, acc13 = {};

    // register double-buffer: loads for k-step i+1 issue before WMMAs of step i
    v16bf a0_n = load_a_frag(A, lda, m0 +  0, 0, lane);
    v16bf a1_n = load_a_frag(A, lda, m0 + 16, 0, lane);
    v16bf b0_n = load_b_frag(Bt, ldb, n0 +  0, 0, lane);
    v16bf b1_n = load_b_frag(Bt, ldb, n0 + 16, 0, lane);
    v16bf b2_n = load_b_frag(Bt, ldb, n0 + 32, 0, lane);
    v16bf b3_n = load_b_frag(Bt, ldb, n0 + 48, 0, lane);

    for (int k0 = 0; k0 < K; k0 += 32) {
        v16bf a0_c = a0_n, a1_c = a1_n;
        v16bf b0_c = b0_n, b1_c = b1_n, b2_c = b2_n, b3_c = b3_n;
        int k1 = k0 + 32;
        if (k1 < K) {
            a0_n = load_a_frag(A, lda, m0 +  0, k1, lane);
            a1_n = load_a_frag(A, lda, m0 + 16, k1, lane);
            b0_n = load_b_frag(Bt, ldb, n0 +  0, k1, lane);
            b1_n = load_b_frag(Bt, ldb, n0 + 16, k1, lane);
            b2_n = load_b_frag(Bt, ldb, n0 + 32, k1, lane);
            b3_n = load_b_frag(Bt, ldb, n0 + 48, k1, lane);
        }
        acc00 = WMMA_BF16(a0_c, b0_c, acc00);
        acc01 = WMMA_BF16(a0_c, b1_c, acc01);
        acc02 = WMMA_BF16(a0_c, b2_c, acc02);
        acc03 = WMMA_BF16(a0_c, b3_c, acc03);
        acc10 = WMMA_BF16(a1_c, b0_c, acc10);
        acc11 = WMMA_BF16(a1_c, b1_c, acc11);
        acc12 = WMMA_BF16(a1_c, b2_c, acc12);
        acc13 = WMMA_BF16(a1_c, b3_c, acc13);
    }

    // C/D layout: lane holds col (lane&15); rows m = r + (lane>=16 ? 8 : 0)
    int n_lane = lane & 15;
    int m_base = (lane & 16) ? 8 : 0;
    v8f accs[2][4] = { { acc00, acc01, acc02, acc03 },
                       { acc10, acc11, acc12, acc13 } };
#pragma unroll
    for (int X = 0; X < 2; ++X) {
#pragma unroll
        for (int t = 0; t < 4; ++t) {
            int n = n0 + t * 16 + n_lane;
            float bv = bias ? bias[n] : 0.0f;
#pragma unroll
            for (int r = 0; r < 8; ++r) {
                int m = m0 + X * 16 + m_base + r;
                float v = accs[X][t][r] + bv;
                if (flags & GF_RELU) v = fmaxf(v, 0.0f);
                int idx = (flags & GF_TRANS) ? (n * ldc + m) : (m * ldc + n);
                if (flags & GF_BF16) ((unsigned short*)C)[idx] = f32_to_bf16_bits(v);
                else                 ((float*)C)[idx] = v;
            }
        }
    }
}

// ---------------------------------------------------------------------------
// Weight preprocessing: f32 W[K][N] row-major -> bf16 Wt[N][K] row-major
// ---------------------------------------------------------------------------
__global__ void convert_transpose(const float* __restrict__ w,
                                  unsigned short* __restrict__ wt, int K, int N) {
    int n = blockIdx.x * 16 + threadIdx.x;
    int k = blockIdx.y * 16 + threadIdx.y;
    if (n < N && k < K) wt[n * K + k] = f32_to_bf16_bits(w[k * N + n]);
}

// Elementwise f32 -> bf16 copy (activation mirrors)
__global__ void convert_rows(const float* __restrict__ in,
                             unsigned short* __restrict__ out, int n) {
    int i = blockIdx.x * 256 + threadIdx.x;
    if (i < n) out[i] = f32_to_bf16_bits(in[i]);
}

// ---------------------------------------------------------------------------
// Row softmax: P[c] = bf16( softmax(S[c] * scale) ). cols <= 2048.
// ---------------------------------------------------------------------------
__global__ __launch_bounds__(256)
void softmax_rows(const float* __restrict__ S, unsigned short* __restrict__ P,
                  int cols, float scale) {
    int row = blockIdx.x, tid = threadIdx.x;
    const float* p = S + row * cols;
    unsigned short* q = P + row * cols;
    __shared__ float red[256];
    float loc[8];
    int cnt = 0;
    float mx = -3.4e38f;
    for (int c = tid; c < cols; c += 256) {
        float v = p[c] * scale;
        loc[cnt++] = v;
        mx = fmaxf(mx, v);
    }
    red[tid] = mx; __syncthreads();
    for (int s = 128; s > 0; s >>= 1) {
        if (tid < s) red[tid] = fmaxf(red[tid], red[tid + s]);
        __syncthreads();
    }
    mx = red[0]; __syncthreads();
    float sum = 0.0f;
    for (int i = 0; i < cnt; ++i) { loc[i] = __expf(loc[i] - mx); sum += loc[i]; }
    red[tid] = sum; __syncthreads();
    for (int s = 128; s > 0; s >>= 1) {
        if (tid < s) red[tid] += red[tid + s];
        __syncthreads();
    }
    float inv = 1.0f / red[0];
    cnt = 0;
    for (int c = tid; c < cols; c += 256) q[c] = f32_to_bf16_bits(loc[cnt++] * inv);
}

// ---------------------------------------------------------------------------
// x = LayerNorm(x + t) * g + b ; also writes bf16 mirror xb.
// ---------------------------------------------------------------------------
__global__ __launch_bounds__(256)
void add_ln(float* __restrict__ x, unsigned short* __restrict__ xb,
            const float* __restrict__ t,
            const float* __restrict__ g, const float* __restrict__ b, int D) {
    int row = blockIdx.x, tid = threadIdx.x;
    float* xr = x + row * D;
    unsigned short* xbr = xb + row * D;
    const float* tr = t + row * D;
    __shared__ float red[256];
    float loc[4];
    int cnt = 0;
    float s0 = 0.0f, s1 = 0.0f;
    for (int c = tid; c < D; c += 256) {
        float v = xr[c] + tr[c];
        loc[cnt++] = v;
        s0 += v; s1 += v * v;
    }
    red[tid] = s0; __syncthreads();
    for (int s = 128; s > 0; s >>= 1) { if (tid < s) red[tid] += red[tid + s]; __syncthreads(); }
    float mean = red[0] / (float)D; __syncthreads();
    red[tid] = s1; __syncthreads();
    for (int s = 128; s > 0; s >>= 1) { if (tid < s) red[tid] += red[tid + s]; __syncthreads(); }
    float var = red[0] / (float)D - mean * mean;
    float inv = rsqrtf(var + 1e-5f);
    cnt = 0;
    for (int c = tid; c < D; c += 256) {
        float v = (loc[cnt++] - mean) * inv * g[c] + b[c];
        xr[c]  = v;
        xbr[c] = f32_to_bf16_bits(v);
    }
}

// ---------------------------------------------------------------------------
// Pooling tail: row L2 norms -> softmax weights -> weighted pooled vector.
// ---------------------------------------------------------------------------
__global__ __launch_bounds__(256)
void row_norms(const float* __restrict__ x, float* __restrict__ nrm, int D) {
    int row = blockIdx.x, tid = threadIdx.x;
    const float* xr = x + row * D;
    __shared__ float red[256];
    float s = 0.0f;
    for (int c = tid; c < D; c += 256) { float v = xr[c]; s += v * v; }
    red[tid] = s; __syncthreads();
    for (int k = 128; k > 0; k >>= 1) { if (tid < k) red[tid] += red[tid + k]; __syncthreads(); }
    if (tid == 0) nrm[row] = sqrtf(red[0]);
}

__global__ __launch_bounds__(256)
void pool_kernel(const float* __restrict__ x, const float* __restrict__ nrm,
                 float* __restrict__ wts, float* __restrict__ pooled, int L, int D) {
    int tid = threadIdx.x;
    __shared__ float red[256];
    float mx = -3.4e38f;
    for (int i = tid; i < L; i += 256) mx = fmaxf(mx, nrm[i]);
    red[tid] = mx; __syncthreads();
    for (int s = 128; s > 0; s >>= 1) { if (tid < s) red[tid] = fmaxf(red[tid], red[tid + s]); __syncthreads(); }
    mx = red[0]; __syncthreads();
    float sum = 0.0f;
    for (int i = tid; i < L; i += 256) { float e = __expf(nrm[i] - mx); wts[i] = e; sum += e; }
    red[tid] = sum; __syncthreads();
    for (int s = 128; s > 0; s >>= 1) { if (tid < s) red[tid] += red[tid + s]; __syncthreads(); }
    float inv = 1.0f / red[0];
    __syncthreads();   // all wts[] visible
    for (int d = tid; d < D; d += 256) {
        float acc = 0.0f;
        for (int m = 0; m < L; ++m) acc += x[m * D + d] * wts[m];
        pooled[d] = acc * inv;
    }
}

__global__ __launch_bounds__(256)
void head_kernel(const float* __restrict__ pooled,
                 const float* __restrict__ w1, const float* __restrict__ b1,
                 const float* __restrict__ w2, const float* __restrict__ b2,
                 float* __restrict__ out) {
    __shared__ float h[256];
    int tid = threadIdx.x;
    float acc = b1[tid];
    for (int d = 0; d < HIDDIM; ++d) acc += pooled[d] * w1[d * 256 + tid];
    h[tid] = fmaxf(acc, 0.0f);
    __syncthreads();
    if (tid < 2) {
        float o = b2[tid];
        for (int j = 0; j < 256; ++j) o += h[j] * w2[j * 2 + tid];
        out[tid] = o;
    }
}

// ---------------------------------------------------------------------------
// Host orchestration
// ---------------------------------------------------------------------------
extern "C" void kernel_launch(void* const* d_in, const int* in_sizes, int n_in,
                              void* d_out, int out_size, void* d_ws, size_t ws_size,
                              hipStream_t stream) {
    (void)in_sizes; (void)n_in; (void)out_size; (void)ws_size;

    const float* trg   = (const float*)d_in[0];
    const float* src   = (const float*)d_in[1];
    const float* ft_w  = (const float*)d_in[2];
    const float* ft_b  = (const float*)d_in[3];
    const float* ln_g  = (const float*)d_in[4];
    const float* ln_b  = (const float*)d_in[5];
    const float* sa_wq = (const float*)d_in[6];  const float* sa_bq = (const float*)d_in[7];
    const float* sa_wk = (const float*)d_in[8];  const float* sa_bk = (const float*)d_in[9];
    const float* sa_wv = (const float*)d_in[10]; const float* sa_bv = (const float*)d_in[11];
    const float* sa_wo = (const float*)d_in[12]; const float* sa_bo = (const float*)d_in[13];
    const float* ea_wq = (const float*)d_in[14]; const float* ea_bq = (const float*)d_in[15];
    const float* ea_wk = (const float*)d_in[16]; const float* ea_bk = (const float*)d_in[17];
    const float* ea_wv = (const float*)d_in[18]; const float* ea_bv = (const float*)d_in[19];
    const float* ea_wo = (const float*)d_in[20]; const float* ea_bo = (const float*)d_in[21];
    const float* pf_w1 = (const float*)d_in[22]; const float* pf_b1 = (const float*)d_in[23];
    const float* pf_w2 = (const float*)d_in[24]; const float* pf_b2 = (const float*)d_in[25];
    const float* fc1_w = (const float*)d_in[26]; const float* fc1_b = (const float*)d_in[27];
    const float* fc2_w = (const float*)d_in[28]; const float* fc2_b = (const float*)d_in[29];

    char* ws = (char*)d_ws;
    size_t off = 0;
    auto alloc = [&](size_t bytes) -> char* {
        char* p = ws + off;
        off += (bytes + 255) & ~(size_t)255;
        return p;
    };

    const int W512  = HIDDIM * HIDDIM;        // 262144
    const int W2048 = HIDDIM * PFDIM;         // 1048576

    // bf16 transposed weights
    unsigned short* ft_t    = (unsigned short*)alloc((size_t)HIDDIM * 64 * 2);
    unsigned short* sa_wq_t = (unsigned short*)alloc((size_t)NLAYER * W512 * 2);
    unsigned short* sa_wk_t = (unsigned short*)alloc((size_t)NLAYER * W512 * 2);
    unsigned short* sa_wv_t = (unsigned short*)alloc((size_t)NLAYER * W512 * 2);
    unsigned short* sa_wo_t = (unsigned short*)alloc((size_t)NLAYER * W512 * 2);
    unsigned short* ea_wq_t = (unsigned short*)alloc((size_t)NLAYER * W512 * 2);
    unsigned short* ea_wk_t = (unsigned short*)alloc((size_t)NLAYER * W512 * 2);
    unsigned short* ea_wv_t = (unsigned short*)alloc((size_t)NLAYER * W512 * 2);
    unsigned short* ea_wo_t = (unsigned short*)alloc((size_t)NLAYER * W512 * 2);
    unsigned short* pf1_t   = (unsigned short*)alloc((size_t)NLAYER * W2048 * 2);
    unsigned short* pf2_t   = (unsigned short*)alloc((size_t)NLAYER * W2048 * 2);

    // activations
    unsigned short* trg_b = (unsigned short*)alloc((size_t)LSEQ * 64 * 2);     // bf16 trg
    unsigned short* src_b = (unsigned short*)alloc((size_t)LSEQ * HIDDIM * 2); // bf16 src
    float*          xbuf  = (float*)alloc((size_t)LSEQ * HIDDIM * 4);          // f32 master x
    unsigned short* xb    = (unsigned short*)alloc((size_t)LSEQ * HIDDIM * 2); // bf16 mirror
    unsigned short* qb    = (unsigned short*)alloc((size_t)LSEQ * HIDDIM * 2); // bf16 Q
    unsigned short* kbuf  = (unsigned short*)alloc((size_t)LSEQ * HIDDIM * 2); // bf16 K
    unsigned short* vTbuf = (unsigned short*)alloc((size_t)HIDDIM * LSEQ * 2); // bf16 V^T
    unsigned short* ob    = (unsigned short*)alloc((size_t)LSEQ * HIDDIM * 2); // bf16 attn O
    float*          tbuf  = (float*)alloc((size_t)LSEQ * HIDDIM * 4);          // f32 sublayer out
    float*          Sbuf  = (float*)alloc((size_t)LSEQ * LSEQ * 4);            // f32 scores
    unsigned short* Pb    = (unsigned short*)alloc((size_t)LSEQ * LSEQ * 2);   // bf16 P / FFN hidden
    float*          nrmb  = (float*)alloc((size_t)LSEQ * 4);
    float*          wtsb  = (float*)alloc((size_t)LSEQ * 4);
    float*          poolb = (float*)alloc((size_t)HIDDIM * 4);

    auto conv = [&](const float* w, unsigned short* wt, int K, int N) {
        dim3 blk(16, 16), grd((N + 15) / 16, (K + 15) / 16);
        convert_transpose<<<grd, blk, 0, stream>>>(w, wt, K, N);
    };
    auto cvrows = [&](const float* in, unsigned short* out, int n) {
        convert_rows<<<(n + 255) / 256, 256, 0, stream>>>(in, out, n);
    };
    auto gemm = [&](const unsigned short* A, int lda, const unsigned short* Bt, int ldb,
                    const float* bias, void* C, int ldc, int M, int N, int K, int flags) {
        dim3 grd(N / 64, (M + 255) / 256);
        gemm_wmma<<<grd, 256, 0, stream>>>(A, lda, Bt, ldb, bias, C, ldc, M, N, K, flags);
    };

    // ---- preprocessing: weights (bf16 transposed) + trg/src bf16 mirrors ----
    conv(ft_w, ft_t, 64, HIDDIM);
    for (int l = 0; l < NLAYER; ++l) {
        conv(sa_wq + l * W512, sa_wq_t + l * W512, HIDDIM, HIDDIM);
        conv(sa_wk + l * W512, sa_wk_t + l * W512, HIDDIM, HIDDIM);
        conv(sa_wv + l * W512, sa_wv_t + l * W512, HIDDIM, HIDDIM);
        conv(sa_wo + l * W512, sa_wo_t + l * W512, HIDDIM, HIDDIM);
        conv(ea_wq + l * W512, ea_wq_t + l * W512, HIDDIM, HIDDIM);
        conv(ea_wk + l * W512, ea_wk_t + l * W512, HIDDIM, HIDDIM);
        conv(ea_wv + l * W512, ea_wv_t + l * W512, HIDDIM, HIDDIM);
        conv(ea_wo + l * W512, ea_wo_t + l * W512, HIDDIM, HIDDIM);
        conv(pf_w1 + l * W2048, pf1_t + l * W2048, HIDDIM, PFDIM);   // [512,2048]->[2048][512]
        conv(pf_w2 + l * W2048, pf2_t + l * W2048, PFDIM, HIDDIM);   // [2048,512]->[512][2048]
    }
    cvrows(trg, trg_b, LSEQ * 64);
    cvrows(src, src_b, LSEQ * HIDDIM);

    // ---- x = trg @ ft_w + ft_b (f32 master), then bf16 mirror ----
    gemm(trg_b, 64, ft_t, 64, ft_b, xbuf, HIDDIM, LSEQ, HIDDIM, 64, 0);
    cvrows(xbuf, xb, LSEQ * HIDDIM);

    const float inv_scale = 0.125f;   // 1/sqrt(64)

    for (int l = 0; l < NLAYER; ++l) {
        const float* g = ln_g + l * HIDDIM;
        const float* b = ln_b + l * HIDDIM;

        // ================= self attention =================
        gemm(xb, HIDDIM, sa_wq_t + l * W512, HIDDIM, sa_bq + l * HIDDIM,
             qb, HIDDIM, LSEQ, HIDDIM, HIDDIM, GF_BF16);
        gemm(xb, HIDDIM, sa_wk_t + l * W512, HIDDIM, sa_bk + l * HIDDIM,
             kbuf, HIDDIM, LSEQ, HIDDIM, HIDDIM, GF_BF16);
        gemm(xb, HIDDIM, sa_wv_t + l * W512, HIDDIM, sa_bv + l * HIDDIM,
             vTbuf, LSEQ, LSEQ, HIDDIM, HIDDIM, GF_BF16 | GF_TRANS);
        for (int h = 0; h < NHEADS; ++h) {
            gemm(qb + h * HDIM, HIDDIM, kbuf + h * HDIM, HIDDIM, nullptr,
                 Sbuf, LSEQ, LSEQ, LSEQ, HDIM, 0);                       // S = Q K^T
            softmax_rows<<<LSEQ, 256, 0, stream>>>(Sbuf, Pb, LSEQ, inv_scale);
            gemm(Pb, LSEQ, vTbuf + h * HDIM * LSEQ, LSEQ, nullptr,
                 ob + h * HDIM, HIDDIM, LSEQ, HDIM, LSEQ, GF_BF16);      // O = P V
        }
        gemm(ob, HIDDIM, sa_wo_t + l * W512, HIDDIM, sa_bo + l * HIDDIM,
             tbuf, HIDDIM, LSEQ, HIDDIM, HIDDIM, 0);
        add_ln<<<LSEQ, 256, 0, stream>>>(xbuf, xb, tbuf, g, b, HIDDIM);

        // ================= cross attention =================
        gemm(xb, HIDDIM, ea_wq_t + l * W512, HIDDIM, ea_bq + l * HIDDIM,
             qb, HIDDIM, LSEQ, HIDDIM, HIDDIM, GF_BF16);
        gemm(src_b, HIDDIM, ea_wk_t + l * W512, HIDDIM, ea_bk + l * HIDDIM,
             kbuf, HIDDIM, LSEQ, HIDDIM, HIDDIM, GF_BF16);
        gemm(src_b, HIDDIM, ea_wv_t + l * W512, HIDDIM, ea_bv + l * HIDDIM,
             vTbuf, LSEQ, LSEQ, HIDDIM, HIDDIM, GF_BF16 | GF_TRANS);
        for (int h = 0; h < NHEADS; ++h) {
            gemm(qb + h * HDIM, HIDDIM, kbuf + h * HDIM, HIDDIM, nullptr,
                 Sbuf, LSEQ, LSEQ, LSEQ, HDIM, 0);
            softmax_rows<<<LSEQ, 256, 0, stream>>>(Sbuf, Pb, LSEQ, inv_scale);
            gemm(Pb, LSEQ, vTbuf + h * HDIM * LSEQ, LSEQ, nullptr,
                 ob + h * HDIM, HIDDIM, LSEQ, HDIM, LSEQ, GF_BF16);
        }
        gemm(ob, HIDDIM, ea_wo_t + l * W512, HIDDIM, ea_bo + l * HIDDIM,
             tbuf, HIDDIM, LSEQ, HIDDIM, HIDDIM, 0);
        add_ln<<<LSEQ, 256, 0, stream>>>(xbuf, xb, tbuf, g, b, HIDDIM);

        // ================= feed forward =================
        gemm(xb, HIDDIM, pf1_t + l * W2048, HIDDIM, pf_b1 + l * PFDIM,
             Pb, PFDIM, LSEQ, PFDIM, HIDDIM, GF_RELU | GF_BF16);         // relu(x W1 + b1)
        gemm(Pb, PFDIM, pf2_t + l * W2048, PFDIM, pf_b2 + l * HIDDIM,
             tbuf, HIDDIM, LSEQ, HIDDIM, PFDIM, 0);
        add_ln<<<LSEQ, 256, 0, stream>>>(xbuf, xb, tbuf, g, b, HIDDIM);
    }

    // ---- norm-softmax pooling + FC head ----
    row_norms<<<LSEQ, 256, 0, stream>>>(xbuf, nrmb, HIDDIM);
    pool_kernel<<<1, 256, 0, stream>>>(xbuf, nrmb, wtsb, poolb, LSEQ, HIDDIM);
    head_kernel<<<1, 256, 0, stream>>>(poolb, fc1_w, fc1_b, fc2_w, fc2_b, (float*)d_out);
}
